// EMAVectorQuantizer_47571057771094
// MI455X (gfx1250) — compile-verified, matrix operand order checked by reading, not run on hardware
//
#include <hip/hip_runtime.h>
#include <hip/hip_bf16.h>

#define BATCH 32
#define DDIM  256
#define HW    1024          // 32*32
#define NROWS 32768         // BATCH*HW
#define KCODE 8192
#define DECAY 0.95f
#define EPS   1e-05f
#define COMMIT 0.25f

typedef __attribute__((ext_vector_type(16))) __bf16 v16bf;
typedef __attribute__((ext_vector_type(8)))  float  v8f;

static __device__ __forceinline__ unsigned short f32_to_bf16_rne(float f) {
    unsigned int u = __float_as_uint(f);
    unsigned int lsb = (u >> 16) & 1u;
    u += 0x7fffu + lsb;
    return (unsigned short)(u >> 16);
}

static __device__ __forceinline__ v16bf frag_cast(uint4 a, uint4 b) {
    union { uint4 u[2]; v16bf v; } t;
    t.u[0] = a; t.u[1] = b;
    return t.v;
}

// Raw LDS byte offset (address space 3) of a __shared__ object.
typedef __attribute__((address_space(3))) const void* lds_cptr;
static __device__ __forceinline__ unsigned lds_addr(const void* p) {
    return (unsigned)(uintptr_t)(lds_cptr)p;
}

// Async global->LDS copy of 32 bytes per lane (2 x b128). ASYNCcnt-tracked;
// the offset immediate advances BOTH the LDS and global address (ISA 08 §4.4).
static __device__ __forceinline__ void stage_tile32(unsigned lds_dst, const uint4* gsrc) {
    asm volatile(
        "global_load_async_to_lds_b128 %0, %1, off\n\t"
        "global_load_async_to_lds_b128 %0, %1, off offset:16"
        :
        : "v"(lds_dst), "v"(gsrc)
        : "memory");
}

static __device__ __forceinline__ void wait_async0() {
    asm volatile("s_wait_asynccnt 0x0" ::: "memory");
}

// ---------------------------------------------------------------------------
// Zero workspace accumulators (dw, counts) and the loss cell of d_out.
// ---------------------------------------------------------------------------
__global__ __launch_bounds__(256) void k_init(float* dw, float* counts, float* o_loss) {
    int tid = blockIdx.x * 256 + threadIdx.x;           // grid covers K*D
    if (tid < KCODE * DDIM) dw[tid] = 0.0f;
    if (tid < KCODE)        counts[tid] = 0.0f;
    if (tid == 0)           o_loss[0] = 0.0f;
}

// ---------------------------------------------------------------------------
// Pack codebook into bf16 WMMA B-fragment order + column norms ||e||^2.
// Layout: uint4 index = ((ct*8 + ks)*32 + lane)*2 + j
// lane: n = ct*16 + (lane&15), half = lane>>4
// slot s (0..15): k = ks*32 + ((s>>3)<<4) + (half<<3) + (s&7)
// ---------------------------------------------------------------------------
__global__ __launch_bounds__(256) void k_pack_b(const float* __restrict__ cb,
                                                uint4* __restrict__ wsB,
                                                float* __restrict__ colNorm) {
    __shared__ float sh[16];
    const int lane = threadIdx.x & 31;
    const int ks   = threadIdx.x >> 5;
    const int ct   = blockIdx.x;
    const int nl   = lane & 15;
    const int half = lane >> 4;
    if (threadIdx.x < 16) sh[threadIdx.x] = 0.0f;
    __syncthreads();

    const float* row = cb + (size_t)(ct * 16 + nl) * DDIM;
    unsigned int w[8];
    float ss = 0.0f;
#pragma unroll
    for (int s = 0; s < 16; ++s) {
        int k = ks * 32 + ((s >> 3) << 4) + (half << 3) + (s & 7);
        float f = row[k];
        ss += f * f;
        unsigned short b = f32_to_bf16_rne(f);
        if (s & 1) w[s >> 1] |= ((unsigned int)b) << 16;
        else       w[s >> 1]  = (unsigned int)b;
    }
    uint4* dst = wsB + ((size_t)(ct * 8 + ks) * 32 + lane) * 2;
    dst[0] = make_uint4(w[0], w[1], w[2], w[3]);
    dst[1] = make_uint4(w[4], w[5], w[6], w[7]);

    atomicAdd(&sh[nl], ss);                 // ds_add_f32
    __syncthreads();
    if (threadIdx.x < 16) colNorm[ct * 16 + threadIdx.x] = sh[threadIdx.x];
}

// ---------------------------------------------------------------------------
// Pack x (NCHW -> NHWC flat rows) into bf16 WMMA A-fragment order.
// Same fragment indexing as B with m = lane&15, row = rt*16 + m, k = channel.
// ---------------------------------------------------------------------------
__global__ __launch_bounds__(256) void k_pack_a(const float* __restrict__ x,
                                                uint4* __restrict__ wsA) {
    const int lane = threadIdx.x & 31;
    const int ks   = threadIdx.x >> 5;
    const int rt   = blockIdx.x;
    const int m    = lane & 15;
    const int half = lane >> 4;
    const int n    = rt * 16 + m;
    const int bi   = n >> 10;
    const int hw   = n & 1023;
    const float* xb = x + (size_t)bi * DDIM * HW + hw;

    unsigned int w[8];
#pragma unroll
    for (int s = 0; s < 16; ++s) {
        int c = ks * 32 + ((s >> 3) << 4) + (half << 3) + (s & 7);
        float f = xb[(size_t)c * HW];
        unsigned short b = f32_to_bf16_rne(f);
        if (s & 1) w[s >> 1] |= ((unsigned int)b) << 16;
        else       w[s >> 1]  = (unsigned int)b;
    }
    uint4* dst = wsA + ((size_t)(rt * 8 + ks) * 32 + lane) * 2;
    dst[0] = make_uint4(w[0], w[1], w[2], w[3]);
    dst[1] = make_uint4(w[4], w[5], w[6], w[7]);
}

// ---------------------------------------------------------------------------
// Main kernel: bf16 WMMA GEMM (flat @ codebook^T) with fused running argmin.
// Block = 8 waves = 128 rows. B tile (16 cols x 256 k, 8 KB) double-buffered
// in LDS via async global->LDS copies (ASYNCcnt), shared by all 8 waves.
// A fragment (16 rows x 256 k) held in VGPRs.
// dist(m,n) ~ ||e_n||^2 - 2 * dot  (||z_m||^2 dropped: constant per row).
// ---------------------------------------------------------------------------
__global__ __launch_bounds__(256) void k_argmin(const uint4* __restrict__ wsA,
                                                const uint4* __restrict__ wsB,
                                                const float* __restrict__ colNorm,
                                                int* __restrict__ idxArr) {
    __shared__ uint4 ldsB[2][512];                    // 2 x 8 KB B fragment tiles
    const int lane = threadIdx.x & 31;
    const int wave = threadIdx.x >> 5;
    const int rt   = blockIdx.x * 8 + wave;           // row tile (16 rows)
    const int nl   = lane & 15;
    const int half = lane >> 4;

    // Preload the wave's full A fragment: 8 k-steps x 32 bytes/lane.
    uint4 afr[8][2];
#pragma unroll
    for (int ks = 0; ks < 8; ++ks) {
        const uint4* src = wsA + ((size_t)(rt * 8 + ks) * 32 + lane) * 2;
        afr[ks][0] = src[0];
        afr[ks][1] = src[1];
    }

    // Per-thread LDS destinations for the async staging (32 B per thread).
    const unsigned ldsDst0 = lds_addr(&ldsB[0][0]) + threadIdx.x * 32u;
    const unsigned ldsDst1 = lds_addr(&ldsB[1][0]) + threadIdx.x * 32u;

    // Prologue: stage tile 0 into buffer 0.
    stage_tile32(ldsDst0, wsB + threadIdx.x * 2);
    wait_async0();
    __syncthreads();

    float bestD[8];
    int   bestN[8];
#pragma unroll
    for (int j = 0; j < 8; ++j) { bestD[j] = 3.402823466e38f; bestN[j] = 0; }

    int buf = 0;
    for (int ct = 0; ct < KCODE / 16; ++ct) {
        // Kick off the async copy of the NEXT tile into the other buffer.
        if (ct + 1 < KCODE / 16)
            stage_tile32(buf ? ldsDst0 : ldsDst1,
                         wsB + (size_t)(ct + 1) * 512 + threadIdx.x * 2);

        // Bulk-load all 8 B fragments for this tile, then WMMA back-to-back.
        const uint4* bsrc = &ldsB[buf][0];
        uint4 bfr[8][2];
#pragma unroll
        for (int ks = 0; ks < 8; ++ks) {
            bfr[ks][0] = bsrc[(ks * 32 + lane) * 2 + 0];
            bfr[ks][1] = bsrc[(ks * 32 + lane) * 2 + 1];
        }

        const int   ncol = ct * 16 + nl;
        const float cn   = colNorm[ncol];

        v8f acc = {};
#pragma unroll
        for (int ks = 0; ks < 8; ++ks) {
            acc = __builtin_amdgcn_wmma_f32_16x16x32_bf16(
                false, frag_cast(afr[ks][0], afr[ks][1]),
                false, frag_cast(bfr[ks][0], bfr[ks][1]),
                (short)0, acc, false, false);
        }
#pragma unroll
        for (int j = 0; j < 8; ++j) {
            float d = fmaf(-2.0f, acc[j], cn);
            if (d < bestD[j] || (d == bestD[j] && ncol < bestN[j])) {
                bestD[j] = d;
                bestN[j] = ncol;
            }
        }

        // Next tile fully in LDS (own asynccnt) + all waves past this tile.
        wait_async0();
        __syncthreads();
        buf ^= 1;
    }

    // Reduce argmin across the 16 lanes holding the same rows.
#pragma unroll
    for (int msk = 1; msk < 16; msk <<= 1) {
#pragma unroll
        for (int j = 0; j < 8; ++j) {
            float od = __shfl_xor(bestD[j], msk, 32);
            int   on = __shfl_xor(bestN[j], msk, 32);
            if (od < bestD[j] || (od == bestD[j] && on < bestN[j])) {
                bestD[j] = od;
                bestN[j] = on;
            }
        }
    }
    if (nl == 0) {
#pragma unroll
        for (int j = 0; j < 8; ++j)
            idxArr[rt * 16 + half * 8 + j] = bestN[j];
    }
}

// ---------------------------------------------------------------------------
// Gather quantized vectors (back to NCHW), commit loss, counts & dw scatter,
// indices output. One block per flat row n, one thread per channel.
// ---------------------------------------------------------------------------
__global__ __launch_bounds__(256) void k_gather(const float* __restrict__ x,
                                                const float* __restrict__ cb,
                                                const int* __restrict__ idxArr,
                                                float* __restrict__ counts,
                                                float* __restrict__ dw,
                                                float* __restrict__ o_loss,
                                                float* __restrict__ o_quant,
                                                float* __restrict__ o_idx) {
    __shared__ float red[256];
    const int n   = blockIdx.x;
    const int c   = threadIdx.x;
    const int idx = idxArr[n];
    const int bi  = n >> 10;
    const int hw  = n & 1023;

    const size_t xoff = ((size_t)bi * DDIM + c) * HW + hw;
    const float xv = x[xoff];
    const float q  = cb[(size_t)idx * DDIM + c];
    o_quant[xoff] = q;                                  // straight-through == quantized
    atomicAdd(&dw[(size_t)idx * DDIM + c], xv);

    float d = q - xv;
    red[c] = d * d;
    __syncthreads();
#pragma unroll
    for (int s = 128; s > 0; s >>= 1) {
        if (c < s) red[c] += red[c + s];
        __syncthreads();
    }
    if (c == 0) {
        atomicAdd(o_loss, red[0] * (COMMIT / (float)((size_t)NROWS * DDIM)));
        atomicAdd(&counts[idx], 1.0f);
        o_idx[n] = (float)idx;
    }
}

// ---------------------------------------------------------------------------
// EMA finalize: new_count (normalized), new_weight, new_codebook.
// ---------------------------------------------------------------------------
__global__ __launch_bounds__(256) void k_ema(const float* __restrict__ ema_count,
                                             const float* __restrict__ ema_weight,
                                             const float* __restrict__ counts,
                                             const float* __restrict__ dw,
                                             float* __restrict__ o_count,
                                             float* __restrict__ o_weight,
                                             float* __restrict__ o_codebook) {
    const int tid = blockIdx.x * 256 + threadIdx.x;     // grid covers K*D
    const int k = tid >> 8;
    const int c = tid & 255;
    const float cnt  = ema_count[k] * DECAY + (1.0f - DECAY) * counts[k];
    const float ncnt = (cnt + EPS) / ((float)NROWS + (float)KCODE * EPS) * (float)NROWS;
    const float w    = ema_weight[tid] * DECAY + (1.0f - DECAY) * dw[tid];
    o_weight[tid]   = w;
    o_codebook[tid] = w / ncnt;
    if (c == 0) o_count[k] = ncnt;
}

// ---------------------------------------------------------------------------
extern "C" void kernel_launch(void* const* d_in, const int* in_sizes, int n_in,
                              void* d_out, int out_size, void* d_ws, size_t ws_size,
                              hipStream_t stream) {
    const float* x          = (const float*)d_in[0];   // [32,256,32,32]
    const float* codebook   = (const float*)d_in[1];   // [8192,256]
    const float* ema_count  = (const float*)d_in[2];   // [8192]
    const float* ema_weight = (const float*)d_in[3];   // [8192,256]

    // d_out layout (flat, reference return order):
    float* o_loss     = (float*)d_out;                               // 1
    float* o_quant    = o_loss + 1;                                  // 8388608
    float* o_idx      = o_quant + (size_t)BATCH * DDIM * HW;         // 32768
    float* o_count    = o_idx + NROWS;                               // 8192
    float* o_weight   = o_count + KCODE;                             // 2097152
    float* o_codebook = o_weight + (size_t)KCODE * DDIM;             // 2097152

    // workspace layout
    char* ws = (char*)d_ws;
    uint4* wsA     = (uint4*)ws;                                     // 16 MB  bf16 A frags
    uint4* wsB     = (uint4*)(ws + (size_t)16 * 1024 * 1024);        //  4 MB  bf16 B frags
    float* colNorm = (float*)(ws + (size_t)20 * 1024 * 1024);        // 32 KB
    int*   idxArr  = (int*)  (ws + (size_t)20 * 1024 * 1024 + 32768);// 128 KB
    float* counts  = (float*)(ws + (size_t)20 * 1024 * 1024 + 32768 + 131072);   // 32 KB
    float* dw      = (float*)(ws + (size_t)20 * 1024 * 1024 + 32768 + 131072 + 32768); // 8 MB

    k_init  <<<(KCODE * DDIM) / 256, 256, 0, stream>>>(dw, counts, o_loss);
    k_pack_b<<<KCODE / 16,           256, 0, stream>>>(codebook, wsB, colNorm);
    k_pack_a<<<NROWS / 16,           256, 0, stream>>>(x, wsA);
    k_argmin<<<NROWS / 128,          256, 0, stream>>>(wsA, wsB, colNorm, idxArr);
    k_gather<<<NROWS,                256, 0, stream>>>(x, codebook, idxArr, counts, dw,
                                                       o_loss, o_quant, o_idx);
    k_ema   <<<(KCODE * DDIM) / 256, 256, 0, stream>>>(ema_count, ema_weight, counts, dw,
                                                       o_count, o_weight, o_codebook);
}
